// LSTMDecoder_5377299055299
// MI455X (gfx1250) — matrix operand order repaired and checked by reading, not compiled, not run on hardware
//
#include <hip/hip_runtime.h>

// Problem sizes (match the reference)
#define LB 128            // batch
#define LS 512            // sequence length
#define LD 1024           // input size
#define LH 1024           // hidden size
#define LG (4 * LH)       // 4096 gate width

typedef __attribute__((ext_vector_type(16))) _Float16 v16h;
typedef __attribute__((ext_vector_type(8)))  _Float16 v8h;
typedef __attribute__((ext_vector_type(8)))  float    v8f;

// ---------------------------------------------------------------------------
// WMMA fragment helpers (16x16x32 f16, wave32; contiguous k order used here, not the ISA table order)
// A 16x32: lane<16 -> row=lane,    K = 0..15 contiguous
//          lane>=16 -> row=lane-16, K = 16..31 contiguous
// B 32x16: lane<16 -> col=lane,    K = 0..15 contiguous
//          lane>=16 -> col=lane-16, K = 16..31 contiguous
// ---------------------------------------------------------------------------
__device__ __forceinline__ v16h make_v16(v8h lo, v8h hi) {
  v16h r;
#pragma unroll
  for (int i = 0; i < 8; ++i) { r[i] = lo[i]; r[i + 8] = hi[i]; }
  return r;
}

// base points at element (row0, k0) of a row-major f16 matrix with leading dim ld
__device__ __forceinline__ v16h load_a_frag(const _Float16* base, int ld) {
  const int lane = threadIdx.x & 31;
  const _Float16* p = base + (size_t)(lane & 15) * ld + ((lane >> 4) << 4);
  v8h lo = *(const v8h*)p;          // K + {0..7}  (or 16..23 for hi lanes)
  v8h hi = *(const v8h*)(p + 8);    // K + {8..15} (or 24..31)
  return make_v16(lo, hi);
}

// base points at element (col0, k0) of w stored row-major (N x K)
__device__ __forceinline__ v16h load_b_frag_global(const _Float16* base, int ld) {
  const int lane = threadIdx.x & 31;
  const _Float16* p = base + (size_t)(lane & 15) * ld + ((lane >> 4) << 4);
  v8h lo = *(const v8h*)p;          // K + {0..7}  (or 16..23 for hi lanes)
  v8h hi = *(const v8h*)(p + 8);    // K + {8..15} (or 24..31)
  return make_v16(lo, hi);
}

// fragment-ordered LDS: 32 lanes x 16 contiguous halfs
__device__ __forceinline__ v16h load_b_frag_lds(const _Float16* lds) {
  const int lane = threadIdx.x & 31;
  const _Float16* p = lds + lane * 16;
  v8h lo = *(const v8h*)p;
  v8h hi = *(const v8h*)(p + 8);
  return make_v16(lo, hi);
}

__device__ __forceinline__ float sigmoidf_(float x) {
  return 1.0f / (1.0f + __expf(-x));
}

// Async global->LDS copy of one 16-byte chunk (per lane), ASYNCcnt-tracked.
__device__ __forceinline__ void async_copy_b128(uint32_t lds_addr,
                                                const _Float16* gsrc) {
  asm volatile("global_load_async_to_lds_b128 %0, %1, off"
               :: "v"(lds_addr), "v"((uint64_t)(uintptr_t)gsrc)
               : "memory");
}

__device__ __forceinline__ void wait_async0() {
  asm volatile("s_wait_asynccnt 0" ::: "memory");
}

// Stage one 128N x 32K B-tile (fragment order) into LDS asynchronously.
// w_base points at (n0, k0) of a row-major (N x K=LD... ld param) matrix.
__device__ __forceinline__ void stage_b_async(const _Float16* w_base, int ld,
                                              _Float16* lds_base, int tid) {
#pragma unroll
  for (int j = 0; j < 2; ++j) {
    int c  = tid + j * 256;
    int st = c >> 6;          // subtile 0..7
    int l  = (c >> 1) & 31;   // lane slot
    int g  = c & 1;           // low/high 8 halfs of the fragment
    const _Float16* src =
        w_base + (size_t)(st * 16 + (l & 15)) * ld + ((l >> 4) << 4) + g * 8;
    uint32_t dst = (uint32_t)(uintptr_t)(lds_base + st * 512 + l * 16 + g * 8);
    async_copy_b128(dst, src);
  }
}

// ---------------------------------------------------------------------------
// Prep kernels
// ---------------------------------------------------------------------------
__global__ void lstm_build_inputs(const float* __restrict__ x,
                                  const float* __restrict__ h0,
                                  _Float16* __restrict__ inp16) {
  size_t idx = (size_t)blockIdx.x * blockDim.x + threadIdx.x;  // < S*B*D
  int d = (int)(idx % LD);
  size_t tb = idx / LD;
  int b = (int)(tb % LB);
  int t = (int)(tb / LB);
  float v = (t == 0) ? h0[(size_t)b * LH + d]
                     : x[(size_t)b * LS * LD + (size_t)(t - 1) * LD + d];
  inp16[idx] = (_Float16)v;
}

__global__ void lstm_f32_to_f16(const float* __restrict__ src,
                                _Float16* __restrict__ dst, int n) {
  int i = blockIdx.x * blockDim.x + threadIdx.x;
  if (i < n) dst[i] = (_Float16)src[i];
}

__global__ void lstm_init_state(const float* __restrict__ h_n,
                                const float* __restrict__ c_n,
                                _Float16* __restrict__ h16,
                                float* __restrict__ c32) {
  int i = blockIdx.x * blockDim.x + threadIdx.x;  // < B*H
  h16[i] = (_Float16)h_n[i];
  c32[i] = c_n[i];
}

// ---------------------------------------------------------------------------
// Phase 1: xg[row][n] = sum_d inp16[row][d]*wih16[n][d] + (b_ih[n]+b_hh[n])
// M = S*B = 65536, N = 4096, K = 1024.
// Workgroup = 8 waves, tile 128M x 128N; wave tile 16M x 128N (8 WMMA accs).
// B tile (128N x 32K f16 = 8KB) double-buffered in LDS via async-to-LDS:
// fetch of tile k+1 overlaps the 8 WMMAs of tile k.
// ---------------------------------------------------------------------------
__global__ __launch_bounds__(256) void lstm_gemm_xg(
    const _Float16* __restrict__ inp16, const _Float16* __restrict__ wih16,
    const float* __restrict__ b_ih, const float* __restrict__ b_hh,
    float* __restrict__ xg) {
  __shared__ _Float16 ldsB[2][8 * 512];  // 2 x 8KB

  const int tid  = threadIdx.x;
  const int lane = tid & 31;
  const int wave = tid >> 5;
  const int n0 = blockIdx.x * 128;
  const int m0 = blockIdx.y * 128 + wave * 16;

  v8f acc[8];
#pragma unroll
  for (int s = 0; s < 8; ++s)
#pragma unroll
    for (int r = 0; r < 8; ++r) acc[s][r] = 0.0f;

  const _Float16* wb = wih16 + (size_t)n0 * LD;

  // prologue: stage tile 0
  stage_b_async(wb, LD, &ldsB[0][0], tid);
  wait_async0();
  __syncthreads();

  for (int k0 = 0; k0 < LD; k0 += 32) {
    const int cur = (k0 >> 5) & 1;
    if (k0 + 32 < LD) {
      stage_b_async(wb + (k0 + 32), LD, &ldsB[cur ^ 1][0], tid);
      __builtin_prefetch(inp16 + (size_t)(m0 + (lane & 15)) * LD + k0 + 32, 0, 3);
    }

    v16h a = load_a_frag(inp16 + (size_t)m0 * LD + k0, LD);
    v16h bfr[8];
#pragma unroll
    for (int s = 0; s < 8; ++s) bfr[s] = load_b_frag_lds(&ldsB[cur][s * 512]);
#pragma unroll
    for (int s = 0; s < 8; ++s)
      acc[s] = __builtin_amdgcn_wmma_f32_16x16x32_f16(
          false, a, false, bfr[s], (short)0, acc[s], false, false);

    wait_async0();     // our async writes for tile k+1 have landed
    __syncthreads();   // everyone done reading tile k / writing tile k+1
  }

  // epilogue: C layout: lane -> col = lane%16, VGPR r -> row = r + 8*(lane/16)
#pragma unroll
  for (int s = 0; s < 8; ++s) {
    int col = n0 + s * 16 + (lane & 15);
    float bias = b_ih[col] + b_hh[col];
#pragma unroll
    for (int r = 0; r < 8; ++r) {
      int row = m0 + r + ((lane >> 4) << 3);
      xg[(size_t)row * LG + col] = acc[s][r] + bias;
    }
  }
}

// ---------------------------------------------------------------------------
// Phase 2: one launch per timestep.
// gates = xg[t] + h @ w_hh^T, fused i/f/g/o nonlinearity + cell update.
// Grid: (H/16 = 64 h-tiles, B/32 = 4 batch blocks) = 256 WGs, 2048 waves.
// Workgroup = 8 waves = 2 batch subtiles x 4-way K split; LDS reduction.
// w_hh (8MB f16) stays L2-resident; B fragments read straight from L2.
// ---------------------------------------------------------------------------
__global__ __launch_bounds__(256) void lstm_step(
    const _Float16* __restrict__ h16_in, _Float16* __restrict__ h16_out,
    float* __restrict__ c32, const float* __restrict__ xg_t,
    const _Float16* __restrict__ whh16, float* __restrict__ out, int t) {
  __shared__ float red[8][4][256];  // [wave][gate][r*32+lane], 32KB

  const int tid  = threadIdx.x;
  const int lane = tid & 31;
  const int wave = tid >> 5;
  const int msub = wave & 1;   // which 16-row batch subtile
  const int ks   = wave >> 1;  // K split 0..3 (256 K each)
  const int h0 = blockIdx.x * 16;
  const int m0 = blockIdx.y * 32 + msub * 16;

  v8f acc[4];
#pragma unroll
  for (int g = 0; g < 4; ++g)
#pragma unroll
    for (int r = 0; r < 8; ++r) acc[g][r] = 0.0f;

  for (int kk = ks * 8; kk < ks * 8 + 8; ++kk) {
    const int k = kk * 32;
    v16h a = load_a_frag(h16_in + (size_t)m0 * LH + k, LH);
    v16h bfr[4];
#pragma unroll
    for (int g = 0; g < 4; ++g)
      bfr[g] = load_b_frag_global(whh16 + (size_t)(g * LH + h0) * LH + k, LH);
#pragma unroll
    for (int g = 0; g < 4; ++g)
      acc[g] = __builtin_amdgcn_wmma_f32_16x16x32_f16(
          false, a, false, bfr[g], (short)0, acc[g], false, false);
  }

  // cross-wave K reduction through LDS
#pragma unroll
  for (int g = 0; g < 4; ++g)
#pragma unroll
    for (int r = 0; r < 8; ++r) red[wave][g][r * 32 + lane] = acc[g][r];
  __syncthreads();

  if (ks == 0) {
    const int h = h0 + (lane & 15);
#pragma unroll
    for (int r = 0; r < 8; ++r) {
      const int b = m0 + r + ((lane >> 4) << 3);  // batch index
      const int e = r * 32 + lane;
      const float* xr = xg_t + (size_t)b * LG;
      float gi = red[msub][0][e] + red[msub + 2][0][e] +
                 red[msub + 4][0][e] + red[msub + 6][0][e] + xr[h];
      float gf = red[msub][1][e] + red[msub + 2][1][e] +
                 red[msub + 4][1][e] + red[msub + 6][1][e] + xr[LH + h];
      float gg = red[msub][2][e] + red[msub + 2][2][e] +
                 red[msub + 4][2][e] + red[msub + 6][2][e] + xr[2 * LH + h];
      float go = red[msub][3][e] + red[msub + 2][3][e] +
                 red[msub + 4][3][e] + red[msub + 6][3][e] + xr[3 * LH + h];
      float iv = sigmoidf_(gi);
      float fv = sigmoidf_(gf);
      float gv = tanhf(gg);
      float ov = sigmoidf_(go);
      const size_t ch = (size_t)b * LH + h;
      float cn = fv * c32[ch] + iv * gv;
      c32[ch] = cn;
      float hn = ov * tanhf(cn);
      out[(size_t)b * (LS * LH) + (size_t)t * LH + h] = hn;
      h16_out[ch] = (_Float16)hn;
    }
  }
}

// ---------------------------------------------------------------------------
// Host launch. Workspace layout (bytes), ~1.14 GiB total required:
//   inp16  @ 0          : S*B*D*2   = 134217728
//   wih16  @ 134217728  : 4H*D*2    = 8388608
//   whh16  @ 142606336  : 4H*H*2    = 8388608
//   h16a   @ 150994944  : B*H*2     = 262144
//   h16b   @ 151257088  : B*H*2     = 262144
//   c32    @ 151519232  : B*H*4     = 524288
//   xg     @ 152043520  : S*B*4H*4  = 1073741824
// ---------------------------------------------------------------------------
extern "C" void kernel_launch(void* const* d_in, const int* in_sizes, int n_in,
                              void* d_out, int out_size, void* d_ws, size_t ws_size,
                              hipStream_t stream) {
  (void)in_sizes; (void)n_in; (void)out_size; (void)ws_size;

  const float* x    = (const float*)d_in[0];
  const float* h_n  = (const float*)d_in[1];
  const float* c_n  = (const float*)d_in[2];
  const float* w_ih = (const float*)d_in[3];
  const float* w_hh = (const float*)d_in[4];
  const float* b_ih = (const float*)d_in[5];
  const float* b_hh = (const float*)d_in[6];
  float* out = (float*)d_out;

  char* ws = (char*)d_ws;
  _Float16* inp16 = (_Float16*)(ws);
  _Float16* wih16 = (_Float16*)(ws + 134217728ull);
  _Float16* whh16 = (_Float16*)(ws + 142606336ull);
  _Float16* h16a  = (_Float16*)(ws + 150994944ull);
  _Float16* h16b  = (_Float16*)(ws + 151257088ull);
  float*    c32   = (float*)   (ws + 151519232ull);
  float*    xg    = (float*)   (ws + 152043520ull);

  // Prep: f16 conversions + state init
  lstm_build_inputs<<<(LS * LB * LD) / 256, 256, 0, stream>>>(x, h_n, inp16);
  lstm_f32_to_f16<<<(LG * LD) / 256, 256, 0, stream>>>(w_ih, wih16, LG * LD);
  lstm_f32_to_f16<<<(LG * LH) / 256, 256, 0, stream>>>(w_hh, whh16, LG * LH);
  lstm_init_state<<<(LB * LH) / 256, 256, 0, stream>>>(h_n, c_n, h16a, c32);

  // Phase 1: big input-projection GEMM
  dim3 g1(LG / 128, (LS * LB) / 128);  // (32, 512)
  lstm_gemm_xg<<<g1, 256, 0, stream>>>(inp16, wih16, b_ih, b_hh, xg);

  // Phase 2: 512 dependent steps, double-buffered hidden state
  _Float16* hin = h16a;
  _Float16* hout = h16b;
  dim3 g2(LH / 16, LB / 32);  // (64, 4) = 256 workgroups
  for (int t = 0; t < LS; ++t) {
    lstm_step<<<g2, 256, 0, stream>>>(hin, hout, c32,
                                      xg + (size_t)t * LB * LG, whh16, out, t);
    _Float16* tmp = hout; hout = hin; hin = tmp;
  }
}
